// SpectralConv1d_3418793968220
// MI455X (gfx1250) — matrix operand order repaired and checked by reading, not compiled, not run on hardware
//
#include <hip/hip_runtime.h>
#include <hip/hip_bf16.h>
#include <math.h>

// B=64, C_IN=64, C_OUT=64, N=8192, M=64
#define N_LEN   8192
#define MODES   64
#define NROWS   4096            // B*C = 64*64 rows for both GEMMs
#define KCOLS   128             // [Re(0..63) | Im(0..63)]
#define KSPLIT  8               // K-segments for stage-1 GEMM
#define TWO_PI_OVER_N 7.6699039394282061e-4f

typedef float  v4f   __attribute__((ext_vector_type(4)));
typedef float  v8f   __attribute__((ext_vector_type(8)));
typedef __bf16 v16bf __attribute__((ext_vector_type(16)));

// ---------------------------------------------------------------------------
// Basis: col c<64 -> cos(2*pi*c*n/N); col c>=64 -> -sin(2*pi*(c-64)*n/N).
// Stored as bf16 hi/lo pairs in BOTH [c][n] (stage-1 B) and [n][c] (stage-3 B)
// layouts so every WMMA B operand is 16 contiguous bf16 per lane.
// ---------------------------------------------------------------------------
__global__ __launch_bounds__(256)
void build_basis(__bf16* __restrict__ Bh, __bf16* __restrict__ Bl,
                 __bf16* __restrict__ Bth, __bf16* __restrict__ Btl) {
    int idx = blockIdx.x * 256 + threadIdx.x;       // c*8192 + n, 1M threads
    int c = idx >> 13;
    int n = idx & (N_LEN - 1);
    int k = c & (MODES - 1);
    int phase = (k * n) & (N_LEN - 1);              // exact integer phase mod N
    float ang = TWO_PI_OVER_N * (float)phase;
    float val = (c < MODES) ? cosf(ang) : -sinf(ang);
    __bf16 h = (__bf16)val;
    __bf16 l = (__bf16)(val - (float)h);
    Bh[idx] = h;
    Bl[idx] = l;
    Bth[(size_t)n * KCOLS + c] = h;
    Btl[(size_t)n * KCOLS + c] = l;
}

// ---------------------------------------------------------------------------
// C[row][col] = sum_k A[row][k] * B[col][k]      (per blockIdx.z K-segment)
// A: fp32 [row][K].  B: bf16 hi/lo [col][K].
// Split precision: acc += Ah*Bh + Ah*Bl + Al*Bh  (3x v_wmma_f32_16x16x32_bf16)
//
// Block = 128 threads (4 waves).  Each wave owns a PRIVATE 16-row stripe and
// 4 column tiles (64 cols).  The 12 WMMAs per k-step are issued round-robin
// across the 4 accumulators so consecutive WMMAs are independent (no D->C
// chain stalls), and they co-execute with the fp32->bf16 conversion VALU.
// ---------------------------------------------------------------------------
__global__ __launch_bounds__(128)
void gemm_abT(const float* __restrict__ A,
              const __bf16* __restrict__ Bh, const __bf16* __restrict__ Bl,
              float* __restrict__ C,
              int ldA, int ksteps, unsigned long long segStride, int ldC) {
    const int tid   = threadIdx.x;
    const int wave  = tid >> 5;
    const int lane  = tid & 31;
    const int laneh = lane >> 4;              // which 16-lane half
    const int lanel = lane & 15;
    const int row0  = (blockIdx.y << 6) + (wave << 4);   // wave-private rows
    const int col0  = blockIdx.x << 6;                   // 4 tiles of 16 cols
    const int kbeg  = blockIdx.z * (ksteps << 5);
    C += (unsigned long long)blockIdx.z * segStride;

    const float*  arow = A  + (size_t)(row0 + lanel) * ldA + kbeg;
    const __bf16* bh0  = Bh + (size_t)(col0 + lanel) * ldA + kbeg;
    const __bf16* bl0  = Bl + (size_t)(col0 + lanel) * ldA + kbeg;
    const size_t  tstr = (size_t)16 * ldA;    // next 16-col tile

    v8f acc0 = {0.f,0.f,0.f,0.f,0.f,0.f,0.f,0.f};
    v8f acc1 = acc0, acc2 = acc0, acc3 = acc0;

    for (int ks = 0; ks < ksteps; ++ks) {
        // ---- A operand: lanes 0-15 hold K 0-15,
        //      lanes 16-31 hold K 16-31 -> one contiguous 16-float chunk.
        const int ka = (ks << 5) + laneh * 16;
        __builtin_prefetch(arow + ka + 128, 0, 1);   // global_prefetch_b8
        v4f a0 = *(const v4f*)(arow + ka);
        v4f a1 = *(const v4f*)(arow + ka + 4);
        v4f a2 = *(const v4f*)(arow + ka + 8);
        v4f a3 = *(const v4f*)(arow + ka + 12);

        // ---- B operands: lanes 0-15 hold K 0-15, lanes 16-31 hold K 16-31,
        //      fixed column -> aligned 32-byte bf16x16 loads (no conversion).
        const int kb = (ks << 5) + laneh * 16;
        v16bf bhv0 = *(const v16bf*)(bh0 + kb);
        v16bf bhv1 = *(const v16bf*)(bh0 + tstr + kb);
        v16bf bhv2 = *(const v16bf*)(bh0 + 2 * tstr + kb);
        v16bf bhv3 = *(const v16bf*)(bh0 + 3 * tstr + kb);
        v16bf blv0 = *(const v16bf*)(bl0 + kb);
        v16bf blv1 = *(const v16bf*)(bl0 + tstr + kb);
        v16bf blv2 = *(const v16bf*)(bl0 + 2 * tstr + kb);
        v16bf blv3 = *(const v16bf*)(bl0 + 3 * tstr + kb);

        v16bf ah, al;
#pragma unroll
        for (int j = 0; j < 4; ++j) {
            float x; __bf16 h;
            x = a0[j]; h = (__bf16)x; ah[j]      = h; al[j]      = (__bf16)(x - (float)h);
            x = a1[j]; h = (__bf16)x; ah[4 + j]  = h; al[4 + j]  = (__bf16)(x - (float)h);
            x = a2[j]; h = (__bf16)x; ah[8 + j]  = h; al[8 + j]  = (__bf16)(x - (float)h);
            x = a3[j]; h = (__bf16)x; ah[12 + j] = h; al[12 + j] = (__bf16)(x - (float)h);
        }

        // ---- 12 WMMAs, round-robin over accumulators: adjacent instructions
        //      are always independent -> no WMMA->WMMA dependency stalls.
        acc0 = __builtin_amdgcn_wmma_f32_16x16x32_bf16(false, ah, false, bhv0, (short)0, acc0, false, false);
        acc1 = __builtin_amdgcn_wmma_f32_16x16x32_bf16(false, ah, false, bhv1, (short)0, acc1, false, false);
        acc2 = __builtin_amdgcn_wmma_f32_16x16x32_bf16(false, ah, false, bhv2, (short)0, acc2, false, false);
        acc3 = __builtin_amdgcn_wmma_f32_16x16x32_bf16(false, ah, false, bhv3, (short)0, acc3, false, false);

        acc0 = __builtin_amdgcn_wmma_f32_16x16x32_bf16(false, ah, false, blv0, (short)0, acc0, false, false);
        acc1 = __builtin_amdgcn_wmma_f32_16x16x32_bf16(false, ah, false, blv1, (short)0, acc1, false, false);
        acc2 = __builtin_amdgcn_wmma_f32_16x16x32_bf16(false, ah, false, blv2, (short)0, acc2, false, false);
        acc3 = __builtin_amdgcn_wmma_f32_16x16x32_bf16(false, ah, false, blv3, (short)0, acc3, false, false);

        acc0 = __builtin_amdgcn_wmma_f32_16x16x32_bf16(false, al, false, bhv0, (short)0, acc0, false, false);
        acc1 = __builtin_amdgcn_wmma_f32_16x16x32_bf16(false, al, false, bhv1, (short)0, acc1, false, false);
        acc2 = __builtin_amdgcn_wmma_f32_16x16x32_bf16(false, al, false, bhv2, (short)0, acc2, false, false);
        acc3 = __builtin_amdgcn_wmma_f32_16x16x32_bf16(false, al, false, bhv3, (short)0, acc3, false, false);
    }

    // ---- C/D layout: VGPR j -> row j (lanes 0-15) / row j+8 (lanes 16-31).
    float* crow = C + (size_t)(row0 + laneh * 8) * ldC + col0 + lanel;
#pragma unroll
    for (int j = 0; j < 8; ++j) {
        crow[(size_t)j * ldC]      = acc0[j];
        crow[(size_t)j * ldC + 16] = acc1[j];
        crow[(size_t)j * ldC + 32] = acc2[j];
        crow[(size_t)j * ldC + 48] = acc3[j];
    }
}

// ---------------------------------------------------------------------------
// Sum the KSPLIT stage-1 partial buffers in a fixed order (deterministic).
// ---------------------------------------------------------------------------
__global__ __launch_bounds__(256)
void reduce_partials(const float* __restrict__ P, float* __restrict__ S1) {
    int idx = blockIdx.x * 256 + threadIdx.x;       // 4096*128 = 524288
    const int seg = NROWS * KCOLS;
    float s = 0.f;
#pragma unroll
    for (int z = 0; z < KSPLIT; ++z) s += P[idx + z * seg];
    S1[idx] = s;
}

// ---------------------------------------------------------------------------
// Per-frequency complex channel mix + irfft scaling folded in:
//   G[row][k]    = ((k==0)?1:2)/N * Re(sum_i Vhat[b,i,k]*W[i,o,k])
//   G[row][64+k] =            2/N * Im(...)     (pairs with the -sin basis)
// ---------------------------------------------------------------------------
__global__ __launch_bounds__(256)
void mix_modes(const float* __restrict__ S1, const float* __restrict__ Wr,
               const float* __restrict__ Wi, float* __restrict__ G) {
    int idx = blockIdx.x * 256 + threadIdx.x;       // 64*64*64 threads
    int k = idx & 63;
    int o = (idx >> 6) & 63;
    int b = idx >> 12;
    const float* vb = S1 + (size_t)b * 64 * KCOLS;
    float R = 0.f, I = 0.f;
    for (int i = 0; i < 64; ++i) {
        float vr = vb[i * KCOLS + k];
        float vi = vb[i * KCOLS + 64 + k];
        float wr = Wr[((i << 6) + o) * 64 + k];
        float wi = Wi[((i << 6) + o) * 64 + k];
        R = fmaf(vr, wr, R); R = fmaf(-vi, wi, R);
        I = fmaf(vr, wi, I); I = fmaf(vi, wr, I);
    }
    const float si = 2.0f / (float)N_LEN;
    const float sr = (k == 0) ? (1.0f / (float)N_LEN) : si;
    size_t row = (size_t)b * 64 + o;
    G[row * KCOLS + k]      = sr * R;
    G[row * KCOLS + 64 + k] = si * I;   // k=0 multiplies -sin(0)=0 downstream
}

extern "C" void kernel_launch(void* const* d_in, const int* in_sizes, int n_in,
                              void* d_out, int out_size, void* d_ws, size_t ws_size,
                              hipStream_t stream) {
    const float* v  = (const float*)d_in[0];   // (64,64,8192)
    const float* wr = (const float*)d_in[1];   // (64,64,64)
    const float* wi = (const float*)d_in[2];   // (64,64,64)
    float* out = (float*)d_out;                // (64,64,8192)

    char* ws = (char*)d_ws;                    // ~28 MB used
    __bf16* Bh  = (__bf16*)(ws);                       // [128][8192] hi   2MB
    __bf16* Bl  = (__bf16*)(ws + (2ull << 20));        // [128][8192] lo   2MB
    __bf16* Bth = (__bf16*)(ws + (4ull << 20));        // [8192][128] hi   2MB
    __bf16* Btl = (__bf16*)(ws + (6ull << 20));        // [8192][128] lo   2MB
    float*  P   = (float*) (ws + (8ull << 20));        // 8 x [4096][128] 16MB
    float*  S1  = (float*) (ws + (24ull << 20));       // [4096][128]      2MB
    float*  G   = (float*) (ws + (26ull << 20));       // [4096][128]      2MB

    // 1) DFT basis (bf16 hi/lo, both layouts)
    build_basis<<<dim3((KCOLS * N_LEN) / 256), 256, 0, stream>>>(Bh, Bl, Bth, Btl);

    // 2) Stage-1 GEMM: Vhat = v (4096x8192) x basis^T -> 8 K-segment partials
    //    grid: 2 col groups x 64 row groups x 8 K-segments = 1024 WGs
    gemm_abT<<<dim3(KCOLS / 64, NROWS / 64, KSPLIT), 128, 0, stream>>>(
        v, Bh, Bl, P, N_LEN, (N_LEN / KSPLIT) / 32,
        (unsigned long long)(NROWS * KCOLS), KCOLS);

    // 3) Deterministic partial reduction
    reduce_partials<<<dim3((NROWS * KCOLS) / 256), 256, 0, stream>>>(P, S1);

    // 4) Complex channel mix per mode + irfft scaling
    mix_modes<<<dim3((64 * 64 * 64) / 256), 256, 0, stream>>>(S1, wr, wi, G);

    // 5) Stage-3 GEMM: out = G (4096x128) x basisT^T (128x8192)
    //    grid: 128 col groups x 64 row groups = 8192 WGs
    gemm_abT<<<dim3(N_LEN / 64, NROWS / 64, 1), 128, 0, stream>>>(
        G, Bth, Btl, out, KCOLS, KCOLS / 32, 0ull, N_LEN);
}
